// Update_coor_9216999817553
// MI455X (gfx1250) — compile-verified, matrix operand order checked by reading, not run on hardware
//
#include <hip/hip_runtime.h>
#include <hip/hip_bf16.h>

#define N_NODES 100000
#define N_EDGES 6400000
#define ATOM_F  128
#define R_CUT   5.0f
#define TILES   (N_NODES / 16)   // 6250 row-tiles of 16

typedef __attribute__((ext_vector_type(16))) __bf16 v16bf;
typedef __attribute__((ext_vector_type(2)))  __bf16 v2bf;
typedef __attribute__((ext_vector_type(8)))  float  v8f;

union Frag { v16bf v; unsigned u[8]; };

// Real fptrunc f32->bf16 (RNE) so the backend can pick native cvt instructions
// instead of the integer round-and-shift chain.
__device__ __forceinline__ unsigned pk(float a, float b) {
    union { v2bf v; unsigned u; } c;
    c.v[0] = (__bf16)a;
    c.v[1] = (__bf16)b;
    return c.u;
}
__device__ __forceinline__ float silu(float x) {
    return x / (1.0f + __expf(-x));
}

// ---------------------------------------------------------------------------
// Node transform: q = silu( silu(x@W1 + b1) @ W2 + b2 ), fused, WMMA bf16.
// One wave handles a 16-row tile. W1 staged in LDS as pre-swizzled bf16
// WMMA-B fragments (4 k-steps x 8 n-tiles x 32 lanes x 8 dwords = 32 KB).
// ---------------------------------------------------------------------------
__global__ __launch_bounds__(256) void node_kernel(
    const float* __restrict__ x,  const float* __restrict__ W1,
    const float* __restrict__ b1, const float* __restrict__ W2,
    const float* __restrict__ b2, float* __restrict__ q)
{
    __shared__ unsigned ldsB[4 * 8 * 32 * 8];   // [ks][nt][lane][vgpr]
    const int tid = threadIdx.x;

    // Cooperative stage: W1 (fp32 row-major [k][n]) -> bf16 B-fragments.
    for (int idx = tid; idx < 4 * 8 * 32 * 8; idx += 256) {
        int v    = idx & 7;
        int lane = (idx >> 3) & 31;
        int nt   = (idx >> 8) & 7;
        int ks   = (idx >> 11) & 3;
        int n  = nt * 16 + (lane & 15);
        int kb = (lane < 16) ? ((v < 4) ? 0 : 16) : ((v < 4) ? 8 : 24);
        int k0 = ks * 32 + kb + 2 * (v & 3);
        ldsB[idx] = pk(W1[k0 * ATOM_F + n], W1[(k0 + 1) * ATOM_F + n]);
    }
    __syncthreads();

    const int lane = tid & 31;
    const int wave = tid >> 5;
    const int tile = blockIdx.x * 8 + wave;
    if (tile >= TILES) return;

    const int m0 = tile * 16;
    const float* xr = x + (size_t)(m0 + (lane & 15)) * ATOM_F;

    v8f acc[8] = {};   // 8 n-tiles of the 16x128 h tile, f32 accumulators

    #pragma unroll
    for (int ks = 0; ks < 4; ++ks) {
        // A fragment: two 8-float runs per lane (consecutive K), cvt->bf16.
        const int kA = ks * 32 + ((lane < 16) ? 0 : 8);
        const int kB = ks * 32 + ((lane < 16) ? 16 : 24);
        float4 f0 = *(const float4*)(xr + kA);
        float4 f1 = *(const float4*)(xr + kA + 4);
        float4 f2 = *(const float4*)(xr + kB);
        float4 f3 = *(const float4*)(xr + kB + 4);
        Frag a;
        a.v[0]  = (__bf16)f0.x; a.v[1]  = (__bf16)f0.y;
        a.v[2]  = (__bf16)f0.z; a.v[3]  = (__bf16)f0.w;
        a.v[4]  = (__bf16)f1.x; a.v[5]  = (__bf16)f1.y;
        a.v[6]  = (__bf16)f1.z; a.v[7]  = (__bf16)f1.w;
        a.v[8]  = (__bf16)f2.x; a.v[9]  = (__bf16)f2.y;
        a.v[10] = (__bf16)f2.z; a.v[11] = (__bf16)f2.w;
        a.v[12] = (__bf16)f3.x; a.v[13] = (__bf16)f3.y;
        a.v[14] = (__bf16)f3.z; a.v[15] = (__bf16)f3.w;

        #pragma unroll
        for (int nt = 0; nt < 8; ++nt) {
            Frag b;
            const unsigned* p = &ldsB[((ks * 8 + nt) * 32 + lane) * 8];
            #pragma unroll
            for (int v = 0; v < 8; ++v) b.u[v] = p[v];   // 2x ds_load_b128
            acc[nt] = __builtin_amdgcn_wmma_f32_16x16x32_bf16(
                false, a.v, false, b.v, (short)0, acc[nt], false, false);
        }
    }

    // Fused layer 2: q[m] = silu( sum_n silu(h[m][n]+b1[n]) * W2[n] + b2 ).
    // C-layout: VGPR i, lane L -> M = i + (L<16?0:8), N = nt*16 + (L&15).
    float part[8];
    #pragma unroll
    for (int i = 0; i < 8; ++i) part[i] = 0.0f;

    #pragma unroll
    for (int nt = 0; nt < 8; ++nt) {
        int n = nt * 16 + (lane & 15);
        float bb = b1[n];
        float ww = W2[n];
        #pragma unroll
        for (int i = 0; i < 8; ++i)
            part[i] += silu(acc[nt][i] + bb) * ww;
    }
    // Reduce over the 16 lanes of each N-half (xor of bits 0..3 stays in-half).
    #pragma unroll
    for (int i = 0; i < 8; ++i) {
        #pragma unroll
        for (int off = 1; off < 16; off <<= 1)
            part[i] += __shfl_xor(part[i], off, 32);
    }
    if ((lane & 15) == 0) {
        float bq = b2[0];
        int mh = m0 + ((lane >> 4) << 3);
        #pragma unroll
        for (int i = 0; i < 8; ++i)
            q[mh + i] = silu(part[i] + bq);
    }
}

// ---------------------------------------------------------------------------
// Edge stage: pure streaming + L2-resident gather/scatter. Memory-bound.
// ---------------------------------------------------------------------------
__global__ __launch_bounds__(256) void zero_kernel(float* __restrict__ out, int n)
{
    int i = blockIdx.x * blockDim.x + threadIdx.x;
    if (i < n) out[i] = 0.0f;
}

__global__ __launch_bounds__(256) void edge_kernel(
    const float* __restrict__ rij, const float* __restrict__ vij,
    const int* __restrict__ src,   const int* __restrict__ dst,
    const float* __restrict__ q,   float* __restrict__ out)
{
    int e = blockIdx.x * blockDim.x + threadIdx.x;
    if (e >= N_EDGES) return;
    float r = rij[e];
    float c = 0.5f * (__cosf(r * (3.14159265358979f / R_CUT)) + 1.0f);
    c = (r < R_CUT) ? c : 0.0f;
    float s = q[src[e]] * c;           // q is 400 KB -> L2-resident gather
    int d = dst[e] * 3;
    size_t eb = (size_t)e * 3;
    float vx = vij[eb + 0], vy = vij[eb + 1], vz = vij[eb + 2];
    __hip_atomic_fetch_add(&out[d + 0], vx * s, __ATOMIC_RELAXED, __HIP_MEMORY_SCOPE_AGENT);
    __hip_atomic_fetch_add(&out[d + 1], vy * s, __ATOMIC_RELAXED, __HIP_MEMORY_SCOPE_AGENT);
    __hip_atomic_fetch_add(&out[d + 2], vz * s, __ATOMIC_RELAXED, __HIP_MEMORY_SCOPE_AGENT);
}

extern "C" void kernel_launch(void* const* d_in, const int* in_sizes, int n_in,
                              void* d_out, int out_size, void* d_ws, size_t ws_size,
                              hipStream_t stream)
{
    const float* x   = (const float*)d_in[0];
    const float* rij = (const float*)d_in[1];
    const float* vij = (const float*)d_in[2];
    const int*   src = (const int*)  d_in[3];
    const int*   dst = (const int*)  d_in[4];
    const float* W1  = (const float*)d_in[5];
    const float* b1  = (const float*)d_in[6];
    const float* W2  = (const float*)d_in[7];
    const float* b2  = (const float*)d_in[8];
    float* out = (float*)d_out;
    float* q   = (float*)d_ws;   // 100000 floats of scratch, fully overwritten

    // 8 waves/block, one 16-row tile per wave.
    node_kernel<<<dim3((TILES + 7) / 8), dim3(256), 0, stream>>>(x, W1, b1, W2, b2, q);
    zero_kernel<<<dim3((3 * N_NODES + 255) / 256), dim3(256), 0, stream>>>(out, 3 * N_NODES);
    edge_kernel<<<dim3((N_EDGES + 255) / 256), dim3(256), 0, stream>>>(rij, vij, src, dst, q, out);
}
